// SCConvNetwork_33492154974470
// MI455X (gfx1250) — matrix operand order repaired and checked by reading, not compiled, not run on hardware
//
#include <hip/hip_runtime.h>
#include <hip/hip_bf16.h>

typedef __attribute__((ext_vector_type(16))) _Float16 v16h;
typedef __attribute__((ext_vector_type(8)))  _Float16 v8h;
typedef __attribute__((ext_vector_type(8)))  float    v8f;

#define WPITCH 264  // padded K pitch (halfs) for LDS-resident W tile: 264*2B=528B
                    // = 132 dwords -> lane-to-lane bank shift of 4 -> conflict-free b128

// ---------------------------------------------------------------------------
// helpers
// ---------------------------------------------------------------------------
__device__ __forceinline__ float sigf(float x) { return 1.0f / (1.0f + __expf(-x)); }

// Stage a 256(K) x 64(N) tile of row-major W (leading dim ldw) into LDS,
// column-major f16: wt[n * WPITCH + k].
__device__ __forceinline__ void load_w_tile(_Float16* wt, const float* __restrict__ W,
                                            int ldw, int n0) {
  for (int idx = threadIdx.x; idx < 64 * 256; idx += blockDim.x) {
    int n = idx & 63;
    int k = idx >> 6;
    wt[n * WPITCH + k] = (_Float16)W[k * ldw + n0 + n];
  }
}

// Pack 16 floats (A fragment) into the 16-bit A-matrix VGPR layout element order.
__device__ __forceinline__ v16h pack16(float4 q0, float4 q1, float4 q2, float4 q3) {
  v16h a;
  a[0]  = (_Float16)q0.x; a[1]  = (_Float16)q0.y; a[2]  = (_Float16)q0.z; a[3]  = (_Float16)q0.w;
  a[4]  = (_Float16)q1.x; a[5]  = (_Float16)q1.y; a[6]  = (_Float16)q1.z; a[7]  = (_Float16)q1.w;
  a[8]  = (_Float16)q2.x; a[9]  = (_Float16)q2.y; a[10] = (_Float16)q2.z; a[11] = (_Float16)q2.w;
  a[12] = (_Float16)q3.x; a[13] = (_Float16)q3.y; a[14] = (_Float16)q3.z; a[15] = (_Float16)q3.w;
  return a;
}

// B fragment: lane (l) wants column col = tile*16 + (l&15), K = k0..k0+7, k0+16..k0+23
// (k0 = kb + 8*(l>>4)), from our column-major LDS tile.
__device__ __forceinline__ v16h load_b(const _Float16* wt, int col, int k0) {
  const v8h* p = (const v8h*)(wt + col * WPITCH + k0);  // 16B aligned
  v8h lo = p[0];
  v8h hi = p[2];
  return __builtin_shufflevector(lo, hi, 0,1,2,3,4,5,6,7,8,9,10,11,12,13,14,15);
}

#define WMMA_F16(A, B, C) \
  __builtin_amdgcn_wmma_f32_16x16x32_f16(false, (A), false, (B), (short)0, (C), false, false)

// ---------------------------------------------------------------------------
// 1) dense GEMM: out[M,256] = x[M,256] @ W[256,256]  (f16 WMMA, f32 accum)
//    grid = (ceil(M/128), 4), block = 256 (8 waves); wave -> 16 rows x 64 cols
// ---------------------------------------------------------------------------
__global__ __launch_bounds__(256) void gemm_xw(const float* __restrict__ x,
                                               const float* __restrict__ W,
                                               float* __restrict__ out, int M) {
  __shared__ _Float16 wt[64 * WPITCH];
  const int n0 = blockIdx.y * 64;
  load_w_tile(wt, W, 256, n0);
  __syncthreads();

  const int tid    = threadIdx.x;
  const int wave   = tid >> 5;
  const int lane   = tid & 31;
  const int lane15 = lane & 15;
  const int hi     = lane >> 4;
  const int m_base = blockIdx.x * 128 + wave * 16;

  int row = m_base + lane15;
  if (row > M - 1) row = M - 1;          // branchless clamp: EXEC stays all-ones
  const float* xrow = x + (size_t)row * 256;

  v8f acc0 = {}, acc1 = {}, acc2 = {}, acc3 = {};

  for (int kb = 0; kb < 256; kb += 32) {
    const float4* ap = (const float4*)(xrow + kb + 8 * hi);   // 32B aligned
    float4 q0 = ap[0], q1 = ap[1], q2 = ap[4], q3 = ap[5];    // K: +0..7, +16..23
    v16h a = pack16(q0, q1, q2, q3);
    const int k0 = kb + 8 * hi;
    acc0 = WMMA_F16(a, load_b(wt,      lane15, k0), acc0);
    acc1 = WMMA_F16(a, load_b(wt, 16 + lane15, k0), acc1);
    acc2 = WMMA_F16(a, load_b(wt, 32 + lane15, k0), acc2);
    acc3 = WMMA_F16(a, load_b(wt, 48 + lane15, k0), acc3);
  }

  #pragma unroll
  for (int r = 0; r < 8; ++r) {
    const int ro = m_base + 8 * hi + r;
    if (ro < M) {
      float* op = out + (size_t)ro * 256 + n0 + lane15;
      op[0]  = acc0[r];
      op[16] = acc1[r];
      op[32] = acc2[r];
      op[48] = acc3[r];
    }
  }
}

// ---------------------------------------------------------------------------
// 2) COO SpMM scatter: acc[rows[e], :] += vals[e] * xw[cols[e], :]
//    one wave per edge; lane handles 8 features
// ---------------------------------------------------------------------------
__global__ __launch_bounds__(256) void spmm_scatter(const int*   __restrict__ rows,
                                                    const int*   __restrict__ cols,
                                                    const float* __restrict__ vals,
                                                    const float* __restrict__ xw,
                                                    float*       __restrict__ acc,
                                                    int nnz) {
  const int e = blockIdx.x * 8 + (threadIdx.x >> 5);
  if (e >= nnz) return;
  const int lane = threadIdx.x & 31;
  const int r = rows[e];
  const int c = cols[e];
  const float v = vals[e];

  const float4* src = (const float4*)(xw + (size_t)c * 256 + lane * 8);
  float4 p0 = src[0], p1 = src[1];
  float* dst = acc + (size_t)r * 256 + lane * 8;
  atomicAdd(dst + 0, v * p0.x);
  atomicAdd(dst + 1, v * p0.y);
  atomicAdd(dst + 2, v * p0.z);
  atomicAdd(dst + 3, v * p0.w);
  atomicAdd(dst + 4, v * p1.x);
  atomicAdd(dst + 5, v * p1.y);
  atomicAdd(dst + 6, v * p1.z);
  atomicAdd(dst + 7, v * p1.w);
}

// ---------------------------------------------------------------------------
// 3) per-bucket row counts (LDS histogram -> 8 global atomics per block)
// ---------------------------------------------------------------------------
__global__ __launch_bounds__(256) void seg_count(const int* __restrict__ belong,
                                                 float* __restrict__ cnt, int n) {
  __shared__ int hist[8];
  if (threadIdx.x < 8) hist[threadIdx.x] = 0;
  __syncthreads();
  const int i = blockIdx.x * 256 + threadIdx.x;
  if (i < n) atomicAdd(&hist[belong[i] & 7], 1);
  __syncthreads();
  if (threadIdx.x < 8) atomicAdd(&cnt[threadIdx.x], (float)hist[threadIdx.x]);
}

// ---------------------------------------------------------------------------
// 4) head: y = sigmoid(acc) @ linW   (bias folded into finalize)
//    pooled per-bucket sums via LDS ds_add_f32, then 512 global atomics/block
// ---------------------------------------------------------------------------
__global__ __launch_bounds__(256) void head_pool(const float* __restrict__ hacc,
                                                 const float* __restrict__ linW,
                                                 const int*   __restrict__ belong,
                                                 float*       __restrict__ P,
                                                 int M) {
  __shared__ _Float16 wt[64 * WPITCH];
  __shared__ float pbuf[8 * 64];
  load_w_tile(wt, linW, 64, 0);
  for (int i = threadIdx.x; i < 512; i += 256) pbuf[i] = 0.0f;
  __syncthreads();

  const int tid    = threadIdx.x;
  const int wave   = tid >> 5;
  const int lane   = tid & 31;
  const int lane15 = lane & 15;
  const int hi     = lane >> 4;
  const int m_base = blockIdx.x * 128 + wave * 16;

  int row = m_base + lane15;
  if (row > M - 1) row = M - 1;
  const float* hrow = hacc + (size_t)row * 256;

  v8f acc0 = {}, acc1 = {}, acc2 = {}, acc3 = {};

  for (int kb = 0; kb < 256; kb += 32) {
    const float4* ap = (const float4*)(hrow + kb + 8 * hi);
    float4 q0 = ap[0], q1 = ap[1], q2 = ap[4], q3 = ap[5];
    q0.x = sigf(q0.x); q0.y = sigf(q0.y); q0.z = sigf(q0.z); q0.w = sigf(q0.w);
    q1.x = sigf(q1.x); q1.y = sigf(q1.y); q1.z = sigf(q1.z); q1.w = sigf(q1.w);
    q2.x = sigf(q2.x); q2.y = sigf(q2.y); q2.z = sigf(q2.z); q2.w = sigf(q2.w);
    q3.x = sigf(q3.x); q3.y = sigf(q3.y); q3.z = sigf(q3.z); q3.w = sigf(q3.w);
    v16h a = pack16(q0, q1, q2, q3);
    const int k0 = kb + 8 * hi;
    acc0 = WMMA_F16(a, load_b(wt,      lane15, k0), acc0);
    acc1 = WMMA_F16(a, load_b(wt, 16 + lane15, k0), acc1);
    acc2 = WMMA_F16(a, load_b(wt, 32 + lane15, k0), acc2);
    acc3 = WMMA_F16(a, load_b(wt, 48 + lane15, k0), acc3);
  }

  #pragma unroll
  for (int r = 0; r < 8; ++r) {
    const int ro = m_base + 8 * hi + r;
    if (ro < M) {
      const int b = belong[ro] & 7;
      atomicAdd(&pbuf[b * 64 +      lane15], acc0[r]);
      atomicAdd(&pbuf[b * 64 + 16 + lane15], acc1[r]);
      atomicAdd(&pbuf[b * 64 + 32 + lane15], acc2[r]);
      atomicAdd(&pbuf[b * 64 + 48 + lane15], acc3[r]);
    }
  }
  __syncthreads();
  for (int i = threadIdx.x; i < 512; i += 256) atomicAdd(&P[i], pbuf[i]);
}

// ---------------------------------------------------------------------------
// 5) finalize: out[b,o] = mean_r( P_r[b,o]/max(cnt_r[b],1) + bias_r[o] )
// ---------------------------------------------------------------------------
__global__ __launch_bounds__(512) void finalize(const float* __restrict__ P,
                                                const float* __restrict__ cnt,
                                                const float* __restrict__ b0,
                                                const float* __restrict__ b1,
                                                const float* __restrict__ b2,
                                                float* __restrict__ out) {
  const int t = threadIdx.x;          // 512 = 8 buckets x 64 outputs
  const int bkt = t >> 6;
  const int o = t & 63;
  const float c0 = fmaxf(cnt[0 + bkt], 1.0f);
  const float c1 = fmaxf(cnt[8 + bkt], 1.0f);
  const float c2 = fmaxf(cnt[16 + bkt], 1.0f);
  const float y0 = P[t]        / c0 + b0[o];
  const float y1 = P[512 + t]  / c1 + b1[o];
  const float y2 = P[1024 + t] / c2 + b2[o];
  out[t] = (y0 + y1 + y2) * (1.0f / 3.0f);
}

__global__ __launch_bounds__(256) void zero_f4(float4* __restrict__ p, long n4) {
  const long i = (long)blockIdx.x * blockDim.x + threadIdx.x;
  if (i < n4) p[i] = make_float4(0.f, 0.f, 0.f, 0.f);
}

// ---------------------------------------------------------------------------
// launch
// ---------------------------------------------------------------------------
extern "C" void kernel_launch(void* const* d_in, const int* in_sizes, int n_in,
                              void* d_out, int out_size, void* d_ws, size_t ws_size,
                              hipStream_t stream) {
  const int N0 = 20000, N1 = 60000, N2 = 40000;

  const float* x0 = (const float*)d_in[0];
  const float* x1 = (const float*)d_in[1];
  const float* x2 = (const float*)d_in[2];

  const int *inc1_r  = (const int*)d_in[3],  *inc1_c  = (const int*)d_in[4];
  const float* inc1_v  = (const float*)d_in[5];
  const int *inc2_r  = (const int*)d_in[6],  *inc2_c  = (const int*)d_in[7];
  const float* inc2_v  = (const float*)d_in[8];
  const int *inc1t_r = (const int*)d_in[9],  *inc1t_c = (const int*)d_in[10];
  const float* inc1t_v = (const float*)d_in[11];
  const int *inc2t_r = (const int*)d_in[12], *inc2t_c = (const int*)d_in[13];
  const float* inc2t_v = (const float*)d_in[14];
  const int *adj0_r  = (const int*)d_in[15], *adj0_c  = (const int*)d_in[16];
  const float* adj0_v  = (const float*)d_in[17];
  const int *adju1_r = (const int*)d_in[18], *adju1_c = (const int*)d_in[19];
  const float* adju1_v = (const float*)d_in[20];
  const int *adjd1_r = (const int*)d_in[21], *adjd1_c = (const int*)d_in[22];
  const float* adjd1_v = (const float*)d_in[23];
  const int *adjd2_r = (const int*)d_in[24], *adjd2_c = (const int*)d_in[25];
  const float* adjd2_v = (const float*)d_in[26];

  const int* bel0 = (const int*)d_in[27];
  const int* bel1 = (const int*)d_in[28];
  const int* bel2 = (const int*)d_in[29];

  const float* W00 = (const float*)d_in[30];
  const float* W10 = (const float*)d_in[31];
  const float* W01 = (const float*)d_in[32];
  const float* W11 = (const float*)d_in[33];
  const float* W21 = (const float*)d_in[34];
  const float* W12 = (const float*)d_in[35];
  const float* W22 = (const float*)d_in[36];
  const float* lin0W = (const float*)d_in[37];
  const float* lin1W = (const float*)d_in[38];
  const float* lin2W = (const float*)d_in[39];
  const float* lin0b = (const float*)d_in[40];
  const float* lin1b = (const float*)d_in[41];
  const float* lin2b = (const float*)d_in[42];

  const int nnz_inc1  = in_sizes[3];
  const int nnz_inc2  = in_sizes[6];
  const int nnz_inc1t = in_sizes[9];
  const int nnz_inc2t = in_sizes[12];
  const int nnz_adj0  = in_sizes[15];
  const int nnz_adju1 = in_sizes[18];
  const int nnz_adjd1 = in_sizes[21];
  const int nnz_adjd2 = in_sizes[24];

  // workspace layout (floats)
  float* ws   = (float*)d_ws;
  float* acc0 = ws;                                  // N0*256
  float* acc1 = acc0 + (size_t)N0 * 256;             // N1*256
  float* acc2 = acc1 + (size_t)N1 * 256;             // N2*256
  float* P    = acc2 + (size_t)N2 * 256;             // 3*512
  float* cnt  = P + 3 * 512;                         // 24
  float* tbuf = cnt + 24;                            // N1*256 (reused GEMM scratch)
  const long nzero = (long)(N0 + N1 + N2) * 256 + 3 * 512 + 24;   // divisible by 4
  const long nz4   = nzero / 4;

  zero_f4<<<(int)((nz4 + 255) / 256), 256, 0, stream>>>((float4*)ws, nz4);

  auto gemm = [&](const float* x, const float* W, float* out, int M) {
    dim3 g((M + 127) / 128, 4);
    gemm_xw<<<g, 256, 0, stream>>>(x, W, out, M);
  };
  auto scat = [&](const int* r, const int* c, const float* v, const float* xw,
                  float* acc, int nnz) {
    spmm_scatter<<<(nnz + 7) / 8, 256, 0, stream>>>(r, c, v, xw, acc, nnz);
  };

  // rank-0 accumulator: adj0 @ (x0 W00)  +  inc1 @ (x1 W10)
  gemm(x0, W00, tbuf, N0);
  scat(adj0_r, adj0_c, adj0_v, tbuf, acc0, nnz_adj0);
  gemm(x1, W10, tbuf, N1);
  scat(inc1_r, inc1_c, inc1_v, tbuf, acc0, nnz_inc1);

  // rank-1: inc1t @ (x0 W01) + (adjd1+adju1) @ (x1 W11) + inc2 @ (x2 W21)
  gemm(x0, W01, tbuf, N0);
  scat(inc1t_r, inc1t_c, inc1t_v, tbuf, acc1, nnz_inc1t);
  gemm(x1, W11, tbuf, N1);
  scat(adjd1_r, adjd1_c, adjd1_v, tbuf, acc1, nnz_adjd1);
  scat(adju1_r, adju1_c, adju1_v, tbuf, acc1, nnz_adju1);
  gemm(x2, W21, tbuf, N2);
  scat(inc2_r, inc2_c, inc2_v, tbuf, acc1, nnz_inc2);

  // rank-2: inc2t @ (x1 W12) + adjd2 @ (x2 W22)
  gemm(x1, W12, tbuf, N1);
  scat(inc2t_r, inc2t_c, inc2t_v, tbuf, acc2, nnz_inc2t);
  gemm(x2, W22, tbuf, N2);
  scat(adjd2_r, adjd2_c, adjd2_v, tbuf, acc2, nnz_adjd2);

  // bucket counts
  seg_count<<<(N0 + 255) / 256, 256, 0, stream>>>(bel0, cnt + 0,  N0);
  seg_count<<<(N1 + 255) / 256, 256, 0, stream>>>(bel1, cnt + 8,  N1);
  seg_count<<<(N2 + 255) / 256, 256, 0, stream>>>(bel2, cnt + 16, N2);

  // heads: sigmoid + @linW + pooled sums
  head_pool<<<(N0 + 127) / 128, 256, 0, stream>>>(acc0, lin0W, bel0, P + 0,    N0);
  head_pool<<<(N1 + 127) / 128, 256, 0, stream>>>(acc1, lin1W, bel1, P + 512,  N1);
  head_pool<<<(N2 + 127) / 128, 256, 0, stream>>>(acc2, lin2W, bel2, P + 1024, N2);

  finalize<<<1, 512, 0, stream>>>(P, cnt, lin0b, lin1b, lin2b, (float*)d_out);
}
